// GPT2Attention_51238959841948
// MI455X (gfx1250) — matrix operand order repaired and checked by reading, not compiled, not run on hardware
//
#include <hip/hip_runtime.h>

// ---------------------------------------------------------------------------
// GPT-2 attention block for MI455X (gfx1250), bf16 WMMA (16x16x32) + f32 acc.
// B=2, S=2048, E=1024, H=16, D=64.
// ---------------------------------------------------------------------------

typedef __bf16 bf16_t;
typedef __attribute__((ext_vector_type(16))) __bf16 v16bf;
typedef __attribute__((ext_vector_type(8)))  float  v8f;

union FragBF {
    uint4 u4[2];   // 2 x 16B = 32B
    v16bf v;       // 16 x bf16 = 32B (8 VGPRs)
};

struct alignas(8) bf4 { bf16_t e[4]; };   // packed 4x bf16 for ds_store_b64

constexpr int B_   = 2;
constexpr int S_   = 2048;
constexpr int E_   = 1024;
constexpr int H_   = 16;
constexpr int D_   = 64;
constexpr int BH_  = B_ * H_;     // 32
constexpr int NQKV = 3 * E_;      // 3072
constexpr int MROW = B_ * S_;     // 4096

// LDS row stride for K=64 tiles (in bf16 elements); 72*2B = 144B (16B-aligned).
constexpr int SK = 72;

// ---------------------------------------------------------------------------
// Kernel 1: qkv = X @ W_attn + b_attn, scattered into Q,K [BH,S,64] bf16 and
//           V transposed [BH,64,S] bf16.
// Block: 256 thr (8 waves), tile 64(M) x 128(N), K chunk = 64 (8 WMMA/wave/step)
// ---------------------------------------------------------------------------
__global__ __launch_bounds__(256)
void qkv_gemm_kernel(const float* __restrict__ X, const float* __restrict__ W,
                     const float* __restrict__ bias,
                     bf16_t* __restrict__ qb, bf16_t* __restrict__ kb,
                     bf16_t* __restrict__ vtb)
{
    __shared__ __align__(16) bf16_t As[64 * SK];    // [m][k]
    __shared__ __align__(16) bf16_t Bs[128 * SK];   // transposed: [n][k]

    const int t     = threadIdx.x;
    const int lane  = t & 31;
    const int wid   = t >> 5;
    const int half  = lane >> 4;     // 0/1
    const int l15   = lane & 15;
    const int waveM = wid >> 2;      // 0..1
    const int waveN = wid & 3;       // 0..3
    const int blockM = blockIdx.y * 64;
    const int blockN = blockIdx.x * 128;

    v8f acc[2][2] = {};

    for (int kk = 0; kk < E_; kk += 64) {
        if (kk) __syncthreads();
        // stage A (64x64 fp32 -> bf16), float4 loads, b64 packed stores
#pragma unroll
        for (int i = 0; i < 4; ++i) {
            int idx = t + i * 256;           // 0..1023 (vec4 units)
            int r  = idx >> 4;               // 0..63
            int c4 = (idx & 15) * 4;         // 0..60
            float4 xv = *(const float4*)&X[(size_t)(blockM + r) * E_ + kk + c4];
            bf4 pk;
            pk.e[0] = (bf16_t)xv.x; pk.e[1] = (bf16_t)xv.y;
            pk.e[2] = (bf16_t)xv.z; pk.e[3] = (bf16_t)xv.w;
            *(bf4*)&As[r * SK + c4] = pk;
        }
        // stage B transposed (64x128 fp32 -> bf16 as [n][k]), float4 loads
#pragma unroll
        for (int i = 0; i < 8; ++i) {
            int idx = t + i * 256;           // 0..2047 (vec4 units)
            int kc = idx >> 5;               // 0..63
            int n4 = (idx & 31) * 4;         // 0..124
            float4 wv = *(const float4*)&W[(size_t)(kk + kc) * NQKV + blockN + n4];
            Bs[(n4 + 0) * SK + kc] = (bf16_t)wv.x;
            Bs[(n4 + 1) * SK + kc] = (bf16_t)wv.y;
            Bs[(n4 + 2) * SK + kc] = (bf16_t)wv.z;
            Bs[(n4 + 3) * SK + kc] = (bf16_t)wv.w;
        }
        if (kk + 64 < E_) {
            __builtin_prefetch(&X[(size_t)(blockM + (t >> 2)) * E_ + kk + 64], 0, 1);
            __builtin_prefetch(&W[(size_t)(kk + 64 + (t >> 7)) * NQKV + blockN + (t & 127)], 0, 1);
        }
        __syncthreads();

        // two K=32 sub-chunks -> 8 WMMAs per wave per barrier
#pragma unroll
        for (int c = 0; c < 2; ++c) {
            v16bf af[2], bfr[2];
#pragma unroll
            for (int i = 0; i < 2; ++i) {
                int row = waveM * 32 + i * 16 + l15;
                FragBF f;
                f.u4[0] = *(const uint4*)&As[row * SK + c * 32 + half * 8];
                f.u4[1] = *(const uint4*)&As[row * SK + c * 32 + 16 + half * 8];
                af[i] = f.v;
                int col = waveN * 32 + i * 16 + l15;
                FragBF g;
                g.u4[0] = *(const uint4*)&Bs[col * SK + c * 32 + half * 8];
                g.u4[1] = *(const uint4*)&Bs[col * SK + c * 32 + 16 + half * 8];
                bfr[i] = g.v;
            }
#pragma unroll
            for (int i = 0; i < 2; ++i)
#pragma unroll
                for (int j = 0; j < 2; ++j)
                    acc[i][j] = __builtin_amdgcn_wmma_f32_16x16x32_bf16(
                        false, af[i], false, bfr[j], (short)0, acc[i][j], false, false);
        }
    }

    // Epilogue: bias + scatter. C layout: VGPR r -> row r+8*half, col = l15.
#pragma unroll
    for (int i = 0; i < 2; ++i) {
        int row0 = blockM + waveM * 32 + i * 16 + 8 * half;
#pragma unroll
        for (int j = 0; j < 2; ++j) {
            int c   = blockN + waveN * 32 + j * 16 + l15;   // 0..3071
            float bs = bias[c];
            int sel = c >> 10;          // 0=Q 1=K 2=V
            int cc  = c & 1023;
            int h   = cc >> 6, d = cc & 63;
#pragma unroll
            for (int r = 0; r < 8; ++r) {
                int rowg = row0 + r;                 // 0..4095
                int b = rowg >> 11, s = rowg & 2047;
                int bh = b * H_ + h;
                bf16_t bv = (bf16_t)(acc[i][j][r] + bs);
                if (sel == 0)      qb[((size_t)bh * S_ + s) * D_ + d] = bv;
                else if (sel == 1) kb[((size_t)bh * S_ + s) * D_ + d] = bv;
                else               vtb[((size_t)bh * D_ + d) * S_ + s] = bv;
            }
        }
    }
}

// ---------------------------------------------------------------------------
// Kernel 2: flash-style causal attention (NO 1/sqrt(D) scaling, per reference).
// Grid (BH, S/128); block 256 = 8 waves; each wave owns a 16-query tile and
// streams 32-key chunks with online softmax. Output ctx bf16 [B,S,E].
// ---------------------------------------------------------------------------
__global__ __launch_bounds__(256)
void attn_kernel(const bf16_t* __restrict__ q, const bf16_t* __restrict__ k,
                 const bf16_t* __restrict__ vt, bf16_t* __restrict__ ctx)
{
    __shared__ __align__(16) bf16_t pbuf[8][16 * 32];   // per-wave P tile

    const int lane = threadIdx.x & 31;
    const int w    = threadIdx.x >> 5;
    const int half = lane >> 4;
    const int l15  = lane & 15;
    const int bh   = blockIdx.x;                 // 0..31
    const int qt   = blockIdx.y * 8 + w;         // 0..127
    const int q0   = qt * 16;

    const bf16_t* qp = q  + (size_t)bh * S_ * D_;
    const bf16_t* kp = k  + (size_t)bh * S_ * D_;
    const bf16_t* vp = vt + (size_t)bh * D_ * S_;

    // Q fragments (A layout), loaded once: d-chunks 0..31 and 32..63
    v16bf qf[2];
#pragma unroll
    for (int c = 0; c < 2; ++c) {
        const bf16_t* base = qp + (size_t)(q0 + l15) * D_ + c * 32;
        FragBF f;
        f.u4[0] = *(const uint4*)(base + half * 8);
        f.u4[1] = *(const uint4*)(base + 16 + half * 8);
        qf[c] = f.v;
    }

    v8f acc[4] = {};
    float mrow[8], lrow[8];
#pragma unroll
    for (int r = 0; r < 8; ++r) { mrow[r] = -1e30f; lrow[r] = 0.0f; }

    const int nk = q0 + 16;                      // causal key bound
    for (int k0 = 0; k0 < nk; k0 += 32) {
        // ----- scores: two 16x16 tiles, each = 2 WMMAs over K=64 -----
        v8f sc[2];
#pragma unroll
        for (int ts = 0; ts < 2; ++ts) {
            v8f c8 = {};
#pragma unroll
            for (int c = 0; c < 2; ++c) {
                const bf16_t* base =
                    kp + (size_t)(k0 + ts * 16 + l15) * D_ + c * 32 + half * 8;
                FragBF g;
                g.u4[0] = *(const uint4*)(base);
                g.u4[1] = *(const uint4*)(base + 16);
                c8 = __builtin_amdgcn_wmma_f32_16x16x32_bf16(
                    false, qf[c], false, g.v, (short)0, c8, false, false);
            }
            sc[ts] = c8;
        }

        // ----- causal mask + online softmax (rows live in 16-lane halves) ----
#pragma unroll
        for (int r = 0; r < 8; ++r) {
            int rowg = q0 + r + 8 * half;
            float s0 = ((k0 + l15)      <= rowg) ? sc[0][r] : -1e30f;
            float s1 = ((k0 + 16 + l15) <= rowg) ? sc[1][r] : -1e30f;
            float mt = fmaxf(s0, s1);
#pragma unroll
            for (int msk = 8; msk >= 1; msk >>= 1)
                mt = fmaxf(mt, __shfl_xor(mt, msk, 32));
            float mnew  = fmaxf(mrow[r], mt);
            float p0    = __expf(s0 - mnew);
            float p1    = __expf(s1 - mnew);
            float scale = __expf(mrow[r] - mnew);
            float ps    = p0 + p1;
#pragma unroll
            for (int msk = 8; msk >= 1; msk >>= 1)
                ps += __shfl_xor(ps, msk, 32);
            lrow[r] = lrow[r] * scale + ps;
            mrow[r] = mnew;
#pragma unroll
            for (int dt = 0; dt < 4; ++dt) acc[dt][r] *= scale;
            // C-layout -> LDS (row r+8*half, cols l15 / 16+l15) as bf16
            pbuf[w][(r + 8 * half) * 32 + l15]      = (bf16_t)p0;
            pbuf[w][(r + 8 * half) * 32 + 16 + l15] = (bf16_t)p1;
        }
        asm volatile("s_wait_dscnt 0x0" ::: "memory");   // LDS store->load

        // reload P in A-fragment layout (16x32)
        FragBF pf;
        pf.u4[0] = *(const uint4*)&pbuf[w][l15 * 32 + half * 8];
        pf.u4[1] = *(const uint4*)&pbuf[w][l15 * 32 + 16 + half * 8];

        // ----- P @ V: 4 WMMAs (one per 16-wide D tile), K=32 keys -----
#pragma unroll
        for (int dt = 0; dt < 4; ++dt) {
            const bf16_t* base =
                vp + (size_t)(dt * 16 + l15) * S_ + k0 + half * 8;
            FragBF g;
            g.u4[0] = *(const uint4*)(base);
            g.u4[1] = *(const uint4*)(base + 16);
            acc[dt] = __builtin_amdgcn_wmma_f32_16x16x32_bf16(
                false, pf.v, false, g.v, (short)0, acc[dt], false, false);
        }
    }

    // Epilogue: normalize, merge heads -> ctx [B,S,E] bf16
    const int b = bh >> 4, h = bh & 15;
#pragma unroll
    for (int r = 0; r < 8; ++r) {
        float inv = 1.0f / lrow[r];
        int s = q0 + r + 8 * half;
#pragma unroll
        for (int dt = 0; dt < 4; ++dt)
            ctx[((size_t)(b * S_ + s)) * E_ + h * D_ + dt * 16 + l15] =
                (bf16_t)(acc[dt][r] * inv);
    }
}

// ---------------------------------------------------------------------------
// Kernel 3: out = ctx(bf16) @ w_proj + b_proj, f32 output.
// A tile (already bf16) staged with CDNA5 async global->LDS DMA
// (global_load_async_to_lds_b128, ASYNCcnt), bypassing VGPRs.
// ---------------------------------------------------------------------------
__global__ __launch_bounds__(256)
void proj_gemm_kernel(const bf16_t* __restrict__ A, const float* __restrict__ W,
                      const float* __restrict__ bias, float* __restrict__ out)
{
    __shared__ __align__(16) bf16_t As[64 * SK];
    __shared__ __align__(16) bf16_t Bs[128 * SK];   // transposed [n][k]

    const int t     = threadIdx.x;
    const int lane  = t & 31;
    const int wid   = t >> 5;
    const int half  = lane >> 4;
    const int l15   = lane & 15;
    const int waveM = wid >> 2;
    const int waveN = wid & 3;
    const int blockM = blockIdx.y * 64;
    const int blockN = blockIdx.x * 128;

    v8f acc[2][2] = {};

    for (int kk = 0; kk < E_; kk += 64) {
        if (kk) __syncthreads();
        // stage A (64x64 bf16) via async global->LDS b128: 2 x 16B per thread
#pragma unroll
        for (int i = 0; i < 2; ++i) {
            int seg = t + i * 256;           // 0..511, each = 8 bf16 = 16B
            int r  = seg >> 3;               // 0..63
            int c8 = (seg & 7) * 8;          // 0..56
            unsigned ldsa = (unsigned)(uintptr_t)&As[r * SK + c8];
            unsigned goff = (unsigned)(((size_t)(blockM + r) * E_ + kk + c8)
                                       * sizeof(bf16_t));
            asm volatile("global_load_async_to_lds_b128 %0, %1, %2"
                         :: "v"(ldsa), "v"(goff), "s"(A) : "memory");
        }
        // stage B transposed (64x128 fp32 -> bf16 as [n][k]), float4 loads
#pragma unroll
        for (int i = 0; i < 8; ++i) {
            int idx = t + i * 256;
            int kc = idx >> 5;               // 0..63
            int n4 = (idx & 31) * 4;         // 0..124
            float4 wv = *(const float4*)&W[(size_t)(kk + kc) * E_ + blockN + n4];
            Bs[(n4 + 0) * SK + kc] = (bf16_t)wv.x;
            Bs[(n4 + 1) * SK + kc] = (bf16_t)wv.y;
            Bs[(n4 + 2) * SK + kc] = (bf16_t)wv.z;
            Bs[(n4 + 3) * SK + kc] = (bf16_t)wv.w;
        }
        if (kk + 64 < E_) {
            __builtin_prefetch(&W[(size_t)(kk + 64 + (t >> 7)) * E_ + blockN + (t & 127)], 0, 1);
        }
        asm volatile("s_wait_asynccnt 0x0" ::: "memory");  // async DMA done
        __syncthreads();

#pragma unroll
        for (int c = 0; c < 2; ++c) {
            v16bf af[2], bfr[2];
#pragma unroll
            for (int i = 0; i < 2; ++i) {
                int row = waveM * 32 + i * 16 + l15;
                FragBF f;
                f.u4[0] = *(const uint4*)&As[row * SK + c * 32 + half * 8];
                f.u4[1] = *(const uint4*)&As[row * SK + c * 32 + 16 + half * 8];
                af[i] = f.v;
                int col = waveN * 32 + i * 16 + l15;
                FragBF g;
                g.u4[0] = *(const uint4*)&Bs[col * SK + c * 32 + half * 8];
                g.u4[1] = *(const uint4*)&Bs[col * SK + c * 32 + 16 + half * 8];
                bfr[i] = g.v;
            }
#pragma unroll
            for (int i = 0; i < 2; ++i)
#pragma unroll
                for (int j = 0; j < 2; ++j)
                    acc[i][j] = __builtin_amdgcn_wmma_f32_16x16x32_bf16(
                        false, af[i], false, bfr[j], (short)0, acc[i][j], false, false);
        }
    }

#pragma unroll
    for (int i = 0; i < 2; ++i) {
        int row0 = blockM + waveM * 32 + i * 16 + 8 * half;
#pragma unroll
        for (int j = 0; j < 2; ++j) {
            int c = blockN + waveN * 32 + j * 16 + l15;
            float bs = bias[c];
#pragma unroll
            for (int r = 0; r < 8; ++r)
                out[(size_t)(row0 + r) * E_ + c] = acc[i][j][r] + bs;
        }
    }
}

// ---------------------------------------------------------------------------
extern "C" void kernel_launch(void* const* d_in, const int* in_sizes, int n_in,
                              void* d_out, int out_size, void* d_ws, size_t ws_size,
                              hipStream_t stream)
{
    (void)in_sizes; (void)n_in; (void)out_size; (void)ws_size;

    const float* X     = (const float*)d_in[0];   // [B,S,E]
    const float* Wqkv  = (const float*)d_in[1];   // [E,3E]
    const float* Bqkv  = (const float*)d_in[2];   // [3E]
    const float* Wproj = (const float*)d_in[3];   // [E,E]
    const float* Bproj = (const float*)d_in[4];   // [E]
    float* out = (float*)d_out;

    // workspace layout (bf16): q[BH,S,64] | k[BH,S,64] | vT[BH,64,S] | ctx[B,S,E]
    // = 4 * 8MB = 32MB
    const size_t qkvElems = (size_t)BH_ * S_ * D_;
    bf16_t* qb  = (bf16_t*)d_ws;
    bf16_t* kb  = qb  + qkvElems;
    bf16_t* vtb = kb  + qkvElems;
    bf16_t* ctx = vtb + qkvElems;

    dim3 blk(256);
    qkv_gemm_kernel<<<dim3(NQKV / 128, MROW / 64), blk, 0, stream>>>(
        X, Wqkv, Bqkv, qb, kb, vtb);
    attn_kernel<<<dim3(BH_, S_ / 128), blk, 0, stream>>>(qb, kb, vtb, ctx);
    proj_gemm_kernel<<<dim3(E_ / 128, MROW / 64), blk, 0, stream>>>(
        ctx, Wproj, Bproj, out);
}
